// DressedQuantumCircuit_566935683393
// MI455X (gfx1250) — matrix operand order repaired and checked by reading, not compiled, not run on hardware
//
#include <hip/hip_runtime.h>
#include <hip/hip_bf16.h>

// ---------------------------------------------------------------------------
// Dressed quantum circuit, fused single kernel for gfx1250 (MI455X, wave32).
//  - pre-layer GEMM via V_WMMA_F32_16X16X32_F16 (16 waves each do one K-chunk)
//    with b128-vectorized A/B tile loads
//  - 10-qubit statevector held entirely in VGPRs: 32 amplitudes/lane (64 VGPRs)
//  - register-bit gates = pure VALU; lane-bit gates = __shfl_xor butterflies
// ---------------------------------------------------------------------------

typedef __attribute__((ext_vector_type(16))) _Float16 v16h;
typedef __attribute__((ext_vector_type(8)))  float    v8f;
typedef __attribute__((ext_vector_type(4)))  float    v4f;

#define DEVFN static __device__ __forceinline__

constexpr int   NQ        = 10;   // qubits
constexpr int   REGS      = 32;   // amplitudes per lane: 1024 / 32 lanes
constexpr int   QDEPTH    = 6;
constexpr int   F_IN      = 512;
constexpr float PI_F      = 3.14159265358979323846f;
constexpr float INV_SQRT2 = 0.70710678118654752440f;

template<int I> struct IC { static constexpr int v = I; };

template<int I, int N, typename F>
DEVFN void static_for(F&& f) {
  if constexpr (I < N) {
    f(IC<I>{});
    static_for<I + 1, N>(static_cast<F&&>(f));
  }
}

DEVFN float sx(float v, int m) { return __shfl_xor(v, m, 32); }

// Real 2x2 gate U=[[u00,u01],[u10,u11]] applied on amplitude bit B.
// B>=5: register-dimension bit (pure VALU). B<5: lane-dimension bit (shfl).
template<int B>
DEVFN void gate2x2(float (&re)[REGS], float (&im)[REGS],
                   float u00, float u01, float u10, float u11, int lane) {
  if constexpr (B >= 5) {
    constexpr int m = 1 << (B - 5);
#pragma unroll
    for (int r = 0; r < REGS; ++r) {
      if ((r & m) == 0) {
        const int r1 = r | m;
        float ar = re[r],  ai = im[r];
        float br = re[r1], bi = im[r1];
        re[r]  = u00 * ar + u01 * br;
        im[r]  = u00 * ai + u01 * bi;
        re[r1] = u10 * ar + u11 * br;
        im[r1] = u10 * ai + u11 * bi;
      }
    }
  } else {
    const int   bit = (lane >> B) & 1;
    const float cs  = bit ? u11 : u00;   // coeff on own amplitude
    const float co  = bit ? u10 : u01;   // coeff on partner amplitude
#pragma unroll
    for (int r = 0; r < REGS; ++r) {
      float pr = sx(re[r], 1 << B);
      float pi = sx(im[r], 1 << B);
      re[r] = cs * re[r] + co * pr;
      im[r] = cs * im[r] + co * pi;
    }
  }
}

// Diagonal phase e^{i phi} on amplitudes with bit B set. (c,s)=(cos,sin)(phi)
template<int B>
DEVFN void phase_gate(float (&re)[REGS], float (&im)[REGS],
                      float c, float s, int lane) {
  if constexpr (B >= 5) {
    constexpr int m = 1 << (B - 5);
#pragma unroll
    for (int r = 0; r < REGS; ++r) {
      if (r & m) {
        float ar = re[r], ai = im[r];
        re[r] = ar * c - ai * s;
        im[r] = ar * s + ai * c;
      }
    }
  } else {
    const int   bit = (lane >> B) & 1;
    const float cc  = bit ? c : 1.0f;
    const float ss  = bit ? s : 0.0f;
#pragma unroll
    for (int r = 0; r < REGS; ++r) {
      float ar = re[r], ai = im[r];
      re[r] = ar * cc - ai * ss;
      im[r] = ar * ss + ai * cc;
    }
  }
}

// CNOT(control wire I, target wire I+1): control bit 9-I, target bit 8-I.
template<int I>
DEVFN void cnot_chain(float (&re)[REGS], float (&im)[REGS], int lane) {
  constexpr int BC = 9 - I;
  constexpr int BT = 8 - I;
  if constexpr (BT >= 5) {
    // both bits in register dimension: pure swap -> free SSA renaming
    constexpr int mc = 1 << (BC - 5), mt = 1 << (BT - 5);
#pragma unroll
    for (int r = 0; r < REGS; ++r) {
      if ((r & mc) && !(r & mt)) {
        const int r1 = r | mt;
        float t;
        t = re[r]; re[r] = re[r1]; re[r1] = t;
        t = im[r]; im[r] = im[r1]; im[r1] = t;
      }
    }
  } else if constexpr (BC >= 5) {
    // control in register dim (bit 5), target = lane bit 4: exchange lanes
    constexpr int mc = 1 << (BC - 5);
#pragma unroll
    for (int r = 0; r < REGS; ++r) {
      if (r & mc) {
        re[r] = sx(re[r], 1 << BT);
        im[r] = sx(im[r], 1 << BT);
      }
    }
  } else {
    // both bits in lane dimension
    const int ctl = (lane >> BC) & 1;
#pragma unroll
    for (int r = 0; r < REGS; ++r) {
      float pr = sx(re[r], 1 << BT);
      float pi = sx(im[r], 1 << BT);
      re[r] = ctl ? pr : re[r];
      im[r] = ctl ? pi : im[r];
    }
  }
}

__global__ __launch_bounds__(512)
void dressed_qc_kernel(const float* __restrict__ feat,      // (B, 512)
                       const float* __restrict__ pre_w,     // (10, 512)
                       const float* __restrict__ pre_b,     // (10,)
                       const float* __restrict__ q_params,  // (60,)
                       const float* __restrict__ post_w,    // (2, 10)
                       const float* __restrict__ post_b,    // (2,)
                       float* __restrict__ out)             // (B, 2)
{
  __shared__ float qacc[16][16];  // WMMA accumulation tile (padded N=16)
  __shared__ float s_qin[16][NQ]; // tanh(pre)*pi/2 per row

  const int lane    = threadIdx.x & 31;
  const int wv      = threadIdx.x >> 5;        // 0..15 waves
  const int rowBase = blockIdx.x * 16;

  if (threadIdx.x < 256) qacc[threadIdx.x >> 4][threadIdx.x & 15] = 0.0f;
  __syncthreads();

  // ---------------- pre-layer GEMM tile via WMMA (one K-chunk per wave) ----
  {
    const int k0   = wv * 32;                  // this wave's K chunk
    const int half = (lane & 16) ? 1 : 0;      // upper lane half?
    const int M    = lane & 15;
    const int N    = lane & 15;

    // A (16x32 f16): lane<16 holds K={0..7,16..23}+k0, lane>=16 adds +8.
    // Two contiguous 8-float runs -> 4x global_load_b128.
    const float* arow = feat + (size_t)(rowBase + M) * F_IN + k0 + half * 8;
    v4f a0 = ((const v4f*)arow)[0];            // K + 0.. 3
    v4f a1 = ((const v4f*)arow)[1];            // K + 4.. 7
    v4f a2 = ((const v4f*)(arow + 16))[0];     // K +16..19
    v4f a3 = ((const v4f*)(arow + 16))[1];     // K +20..23

    // B (32x16 f16) = pre_w^T: same K order as A, two 8-float runs -> 4x b128.
    // Clamp row for N>=10 (stay in bounds) and zero via mask multiply.
    const float zmask = (N < NQ) ? 1.0f : 0.0f;
    const float* brow = pre_w + (size_t)(N < NQ ? N : 0) * F_IN + k0 + half * 8;
    v4f b0 = ((const v4f*)brow)[0] * zmask;
    v4f b1 = ((const v4f*)brow)[1] * zmask;
    v4f b2 = ((const v4f*)(brow + 16))[0] * zmask;
    v4f b3 = ((const v4f*)(brow + 16))[1] * zmask;

    v16h a, bm;
#pragma unroll
    for (int e = 0; e < 4; ++e) {
      a[e]      = (_Float16)a0[e];
      a[e + 4]  = (_Float16)a1[e];
      a[e + 8]  = (_Float16)a2[e];
      a[e + 12] = (_Float16)a3[e];
      bm[e]      = (_Float16)b0[e];
      bm[e + 4]  = (_Float16)b1[e];
      bm[e + 8]  = (_Float16)b2[e];
      bm[e + 12] = (_Float16)b3[e];
    }

    v8f c = {};
    c = __builtin_amdgcn_wmma_f32_16x16x32_f16(
            /*neg_a=*/false, a, /*neg_b=*/false, bm,
            /*c_mod=*/(short)0, c, /*reuse_a=*/false, /*reuse_b=*/false);

    // D layout: VGPR r -> M = r + 8*half, N = lane&15. Combine 16 K-partials.
#pragma unroll
    for (int r = 0; r < 8; ++r)
      atomicAdd(&qacc[r + half * 8][N], c[r]);
  }
  __syncthreads();

  if (threadIdx.x < 16 * NQ) {
    const int M = threadIdx.x / NQ, j = threadIdx.x % NQ;
    s_qin[M][j] = tanhf(qacc[M][j] + pre_b[j]) * (PI_F * 0.5f);
  }
  __syncthreads();

  // ---------------- per-wave 10-qubit statevector simulation --------------
  // amplitude index a = r*32 + lane ; wire j <-> bit (9-j)
  float re[REGS], im[REGS];
#pragma unroll
  for (int r = 0; r < REGS; ++r) { re[r] = 0.0f; im[r] = 0.0f; }
  re[0] = (lane == 0) ? 1.0f : 0.0f;

  float x[NQ];
#pragma unroll
  for (int i = 0; i < NQ; ++i) x[i] = s_qin[wv][i];

  // feature-map phase angles (identical for both reps)
  float c1[NQ], s1[NQ], c2[NQ - 1], s2[NQ - 1];
#pragma unroll
  for (int i = 0; i < NQ; ++i) {
    const float a = 2.0f * x[i];
    c1[i] = cosf(a); s1[i] = sinf(a);
  }
#pragma unroll
  for (int i = 0; i < NQ - 1; ++i) {
    const float a = 2.0f * (PI_F - x[i]) * (PI_F - x[i + 1]);
    c2[i] = cosf(a); s2[i] = sinf(a);
  }

  // ZZFeatureMap, reps = 2 (same angles each rep -> runtime loop, small code)
#pragma unroll 1
  for (int rep = 0; rep < 2; ++rep) {
    static_for<0, NQ>([&](auto t) {
      constexpr int i = decltype(t)::v;
      gate2x2<9 - i>(re, im, INV_SQRT2, INV_SQRT2, INV_SQRT2, -INV_SQRT2, lane);
      phase_gate<9 - i>(re, im, c1[i], s1[i], lane);
    });
    static_for<0, NQ - 1>([&](auto t) {
      constexpr int i = decltype(t)::v;
      cnot_chain<i>(re, im, lane);
      phase_gate<8 - i>(re, im, c2[i], s2[i], lane);
    });
  }

  // RealAmplitudes: 6 layers of RY + CNOT chain
#pragma unroll 1
  for (int l = 0; l < QDEPTH; ++l) {
    const float* qp = q_params + l * NQ;   // uniform -> scalar loads
    float cw[NQ], sw[NQ];
#pragma unroll
    for (int i = 0; i < NQ; ++i) {
      const float th = 0.5f * qp[i];
      cw[i] = cosf(th); sw[i] = sinf(th);
    }
    static_for<0, NQ>([&](auto t) {
      constexpr int i = decltype(t)::v;
      gate2x2<9 - i>(re, im, cw[i], -sw[i], sw[i], cw[i], lane);
    });
    static_for<0, NQ - 1>([&](auto t) {
      constexpr int i = decltype(t)::v;
      cnot_chain<i>(re, im, lane);
    });
  }

  // ---------------- PauliZ expectations + post-layer ----------------------
  float ez[NQ];
#pragma unroll
  for (int j = 0; j < NQ; ++j) ez[j] = 0.0f;

  float lsign[5];
#pragma unroll
  for (int j = 5; j < NQ; ++j)
    lsign[j - 5] = ((lane >> (9 - j)) & 1) ? -1.0f : 1.0f;

#pragma unroll
  for (int r = 0; r < REGS; ++r) {
    const float p = re[r] * re[r] + im[r] * im[r];
#pragma unroll
    for (int j = 0; j < 5; ++j) {            // wires 0..4: register bits
      if ((r >> (4 - j)) & 1) ez[j] -= p; else ez[j] += p;
    }
#pragma unroll
    for (int j = 5; j < NQ; ++j)             // wires 5..9: lane bits
      ez[j] += p * lsign[j - 5];
  }

  float o0 = 0.0f, o1 = 0.0f;
#pragma unroll
  for (int j = 0; j < NQ; ++j) {
    o0 += ez[j] * post_w[j];
    o1 += ez[j] * post_w[NQ + j];
  }
#pragma unroll
  for (int off = 16; off > 0; off >>= 1) {
    o0 += sx(o0, off);
    o1 += sx(o1, off);
  }
  if (lane == 0) {
    const int row = rowBase + wv;
    out[row * 2 + 0] = o0 + post_b[0];
    out[row * 2 + 1] = o1 + post_b[1];
  }
}

extern "C" void kernel_launch(void* const* d_in, const int* in_sizes, int n_in,
                              void* d_out, int out_size, void* d_ws, size_t ws_size,
                              hipStream_t stream) {
  (void)n_in; (void)out_size; (void)d_ws; (void)ws_size;
  const float* feat     = (const float*)d_in[0];
  const float* pre_w    = (const float*)d_in[1];
  const float* pre_b    = (const float*)d_in[2];
  const float* q_params = (const float*)d_in[3];
  const float* post_w   = (const float*)d_in[4];
  const float* post_b   = (const float*)d_in[5];
  float* out            = (float*)d_out;

  const int B = in_sizes[0] / F_IN;          // 16384
  dim3 grid(B / 16), block(512);             // 16 waves = 16 rows per block
  dressed_qc_kernel<<<grid, block, 0, stream>>>(
      feat, pre_w, pre_b, q_params, post_w, post_b, out);
}
